// VqVAE_51634096833039
// MI455X (gfx1250) — matrix operand order repaired and checked by reading, not compiled, not run on hardware
//
#include <hip/hip_runtime.h>
#include <hip/hip_bf16.h>

// Problem constants (from reference)
#define B_   512
#define F_   256
#define FTS_ 255
#define U0_  512
#define U1_  256
#define DIM_ 64
#define EMB_ 512
#define BT   64          // batch-tile rows per workgroup (4 M-tiles of 16)
#define PANEL_ELEMS (32 * 512)   // one 32-col x 512-K bf16 weight panel (32 KB)

typedef __attribute__((ext_vector_type(16))) __bf16 v16bf;
typedef __attribute__((ext_vector_type(8)))  float  v8f;
typedef __attribute__((ext_vector_type(4)))  int    v4i;

union Frag { v16bf v; uint4 q[2]; };

#if __has_builtin(__builtin_amdgcn_global_load_async_to_lds_b128) && \
    __has_builtin(__builtin_amdgcn_s_wait_asynccnt)
#define HAVE_ASYNC 1
typedef __attribute__((address_space(1))) v4i* as1_v4i;   // global int4*
typedef __attribute__((address_space(3))) v4i* as3_v4i;   // LDS int4*
#else
#define HAVE_ASYNC 0
#endif

__device__ __forceinline__ unsigned short f2bf(float f) {
  union { float f; unsigned u; } x; x.f = f;
  unsigned r = x.u + 0x7FFFu + ((x.u >> 16) & 1u);   // round-to-nearest-even
  return (unsigned short)(r >> 16);
}

// ---------------------------------------------------------------------------
// Prep: transpose+convert weights f32 [F][I][O] -> bf16 [F][Opad][Kpad]
// (B^T layout: each WMMA B-fragment lane reads two 16-byte runs of K, 16 elements apart)
// ---------------------------------------------------------------------------
__global__ void prep_w(const float* __restrict__ w, unsigned short* __restrict__ out,
                       int I, int O, int Kpad, int Opad, long total) {
  long idx = (long)blockIdx.x * blockDim.x + threadIdx.x;
  if (idx >= total) return;
  int  k = (int)(idx % Kpad);
  long t = idx / Kpad;
  int  o = (int)(t % Opad);
  int  f = (int)(t / Opad);
  float v = 0.f;
  if (k < I && o < O) v = w[((long)f * I + k) * O + o];
  out[idx] = f2bf(v);
}

// codebook [D][E] f32 -> cbT [E][D] bf16, plus cnorm[e] = ||c_e||^2
__global__ void prep_cb(const float* __restrict__ cb, unsigned short* __restrict__ cbT,
                        float* __restrict__ cnorm) {
  int e = blockIdx.x * blockDim.x + threadIdx.x;
  if (e >= EMB_) return;
  float s = 0.f;
  for (int d = 0; d < DIM_; ++d) {
    float v = cb[(long)d * EMB_ + e];
    s += v * v;
    cbT[(long)e * DIM_ + d] = f2bf(v);
  }
  cnorm[e] = s;
}

// ---------------------------------------------------------------------------
// Stage one 32-column weight panel [32][Kpad] bf16 into LDS.
// Async (ASYNCcnt-tracked) DMA when the toolchain exposes the gfx1250 builtin.
// ---------------------------------------------------------------------------
__device__ __forceinline__ void stage_panel(const unsigned short* __restrict__ wt,
                                            unsigned short* __restrict__ sW,
                                            int colBase, int Kpad) {
  const int tid   = threadIdx.x;
  const int nb128 = (32 * Kpad) >> 3;          // 16B chunks in the panel
  for (int i = tid; i < nb128; i += 256) {
    int elem = i << 3;
    int col  = elem / Kpad;
    int k    = elem - col * Kpad;
    const unsigned short* g = wt + (size_t)(colBase + col) * Kpad + k;
#if HAVE_ASYNC
    __builtin_amdgcn_global_load_async_to_lds_b128(
        (as1_v4i)(unsigned short*)g, (as3_v4i)(sW + elem), 0, 0);
#else
    *(uint4*)(sW + elem) = *(const uint4*)g;
#endif
  }
}

__device__ __forceinline__ void wait_async_lds() {
#if HAVE_ASYNC
  __builtin_amdgcn_s_wait_asynccnt(0);
#endif
}

// ---------------------------------------------------------------------------
// One dense bank layer on a 64-row tile: out = act(in @ W^T + b)
//   inA : LDS bf16 [BT][Kpad]     wt : global bf16 [Opad][Kpad] (one feature)
//   sW  : LDS double-buffer, 2 panels of [32][Kpad]
//   outL: LDS bf16 [BT][Opad]     actKind: 0=relu 1=sigmoid
// 8 waves: wave = (nhalf<<2)|mtile. Panel pair p covers cols [32p, 32p+32);
// wave's chunk = 32p + 16*nhalf. Weight DMA for pair p+1 overlaps WMMA on p.
// ---------------------------------------------------------------------------
__device__ __forceinline__ void dense_bank(
    const unsigned short* __restrict__ inA, unsigned short* __restrict__ outL,
    unsigned short* __restrict__ sW,
    const unsigned short* __restrict__ wt,  const float* __restrict__ bias,
    int Kpad, int O, int Opad, int actKind,
    bool toGlobal, float* __restrict__ gout, int gstride) {
  const int tid  = threadIdx.x;
  const int lane = tid & 31;
  const int wv   = tid >> 5;
  const int mt   = wv & 3;          // M tile (rows mt*16 .. mt*16+15)
  const int nh   = wv >> 2;         // N half within the 32-col panel
  const int m    = lane & 15;
  const int kh   = lane >> 4;       // which K-half this lane holds
  const int npairs = Opad >> 5;

  const unsigned short* arow = inA + (size_t)(mt * 16 + m) * Kpad + kh * 8;

  stage_panel(wt, sW, 0, Kpad);                         // prologue DMA
  for (int p = 0; p < npairs; ++p) {
    wait_async_lds();
    __syncthreads();                                    // panel p resident
    if (p + 1 < npairs)                                 // overlap next DMA
      stage_panel(wt, sW + ((p + 1) & 1) * PANEL_ELEMS, (p + 1) * 32, Kpad);

    const unsigned short* buf = sW + (p & 1) * PANEL_ELEMS;
    const int n = p * 32 + nh * 16 + m;                 // output column
    v8f acc = {0.f, 0.f, 0.f, 0.f, 0.f, 0.f, 0.f, 0.f};
    const unsigned short* brow = buf + (size_t)(nh * 16 + m) * Kpad + kh * 8;
    for (int k0 = 0; k0 < Kpad; k0 += 32) {
      Frag a, b;
      a.q[0] = *(const uint4*)(arow + k0);        // A: K = k0+kh*8   .. +7
      a.q[1] = *(const uint4*)(arow + k0 + 16);   //    K = k0+kh*8+16.. +7
      b.q[0] = *(const uint4*)(brow + k0);        // B: K = k0+kh*8   .. +7
      b.q[1] = *(const uint4*)(brow + k0 + 16);   //    K = k0+kh*8+16.. +7
      acc = __builtin_amdgcn_wmma_f32_16x16x32_bf16(false, a.v, false, b.v,
                                                    (short)0, acc, false, false);
    }
    float bv = (n < O) ? bias[n] : 0.f;
#pragma unroll
    for (int r = 0; r < 8; ++r) {
      int   mm = kh * 8 + r;                      // C layout: row = 8*kh + r
      float v  = acc[r] + bv;
      v = (actKind == 0) ? fmaxf(v, 0.f) : (1.f / (1.f + expf(-v)));
      if (!toGlobal) {
        outL[(size_t)(mt * 16 + mm) * Opad + n] = f2bf(v);
      } else if (n < O) {
        gout[(size_t)(mt * 16 + mm) * gstride + n] = v;
      }
    }
    __syncthreads();            // done reading panel p before it is restaged
  }
}

// ---------------------------------------------------------------------------
// Fused VQ-VAE forward: one workgroup per (feature f, 64-row batch tile).
// All activations live in LDS; weights are async-DMA'd into an LDS panel
// double-buffer shared by all 8 waves (4x less L2 traffic than per-wave loads).
// ---------------------------------------------------------------------------
__global__ __launch_bounds__(256, 1)
void vqvae_fused(const float* __restrict__ x,
                 const float* __restrict__ b1, const float* __restrict__ b2,
                 const float* __restrict__ b3, const float* __restrict__ b4,
                 const float* __restrict__ b5, const float* __restrict__ b6,
                 const unsigned short* __restrict__ wt1, const unsigned short* __restrict__ wt2,
                 const unsigned short* __restrict__ wt3, const unsigned short* __restrict__ wt4,
                 const unsigned short* __restrict__ wt5, const unsigned short* __restrict__ wt6,
                 const unsigned short* __restrict__ cbT, const float* __restrict__ cnorm,
                 float* __restrict__ out) {
  __shared__ __align__(16) unsigned short sA[BT * 512];        // 64 KB
  __shared__ __align__(16) unsigned short sB[BT * 512];        // 64 KB
  __shared__ __align__(16) unsigned short sW[2 * PANEL_ELEMS]; // 64 KB DMA buf
  __shared__ float bestV[BT * 32];                             // 8 KB
  __shared__ int   bestI[BT * 32];                             // 8 KB
  __shared__ int   qIdx[BT];

  const int tid   = threadIdx.x;
  const int bidB  = blockIdx.x & 7;       // B_/BT == 8
  const int f     = blockIdx.x >> 3;
  const int bbase = bidB * BT;

  // Stage x tile (transposed view: rows of x[b, f, :]) into sA as bf16 [BT][256]
  for (int idx = tid; idx < BT * 256; idx += 256) {
    int   i = idx >> 8;
    int   k = idx & 255;
    float v = (k < FTS_) ? x[((size_t)(bbase + i) * F_ + f) * FTS_ + k] : 0.f;
    sA[idx] = f2bf(v);
  }
  __syncthreads();

  dense_bank(sA, sB, sW, wt1 + (size_t)f * 512 * 256, b1 + (size_t)f * U0_,
             256, U0_, 512, 0, false, nullptr, 0);
  __syncthreads();
  dense_bank(sB, sA, sW, wt2 + (size_t)f * 256 * 512, b2 + (size_t)f * U1_,
             512, U1_, 256, 0, false, nullptr, 0);
  __syncthreads();
  dense_bank(sA, sB, sW, wt3 + (size_t)f * 64 * 256, b3 + (size_t)f * DIM_,
             256, DIM_, 64, 0, false, nullptr, 0);     // z in sB, [BT][64]
  __syncthreads();

  // ---- VQ: argmin_e ||z-c_e||^2 == argmax_e (2 z.c_e - ||c_e||^2) via WMMA ----
  {
    const int lane = tid & 31, wv = tid >> 5, mt = wv & 3, nh = wv >> 2;
    const int m = lane & 15, kh = lane >> 4;
    float bv[8]; int bi[8];
#pragma unroll
    for (int r = 0; r < 8; ++r) { bv[r] = -3.4e38f; bi[r] = 0; }
    const unsigned short* arow = sB + (size_t)(mt * 16 + m) * 64 + kh * 8;
    for (int nc = nh; nc < (EMB_ >> 4); nc += 2) {
      const int e = (nc << 4) + m;
      v8f acc = {0.f, 0.f, 0.f, 0.f, 0.f, 0.f, 0.f, 0.f};
      const unsigned short* brow = cbT + (size_t)e * DIM_ + kh * 8;
#pragma unroll
      for (int k0 = 0; k0 < DIM_; k0 += 32) {
        Frag a, b;
        a.q[0] = *(const uint4*)(arow + k0);
        a.q[1] = *(const uint4*)(arow + k0 + 16);
        b.q[0] = *(const uint4*)(brow + k0);
        b.q[1] = *(const uint4*)(brow + k0 + 16);
        acc = __builtin_amdgcn_wmma_f32_16x16x32_bf16(false, a.v, false, b.v,
                                                      (short)0, acc, false, false);
      }
      float cn = cnorm[e];
#pragma unroll
      for (int r = 0; r < 8; ++r) {
        float v = 2.f * acc[r] - cn;
        if (v > bv[r] || (v == bv[r] && e < bi[r])) { bv[r] = v; bi[r] = e; }
      }
    }
    const int slot = m + 16 * nh;
#pragma unroll
    for (int r = 0; r < 8; ++r) {
      int row = mt * 16 + kh * 8 + r;
      bestV[row * 32 + slot] = bv[r];
      bestI[row * 32 + slot] = bi[r];
    }
  }
  __syncthreads();
  if (tid < BT) {       // per-row reduction over 32 candidate slots
    float best = -3.4e38f; int bidx = 0;
    for (int s = 0; s < 32; ++s) {
      float v = bestV[tid * 32 + s];
      int   i = bestI[tid * 32 + s];
      if (v > best || (v == best && i < bidx)) { best = v; bidx = i; }
    }
    qIdx[tid] = bidx;
  }
  __syncthreads();
  // gather quantized rows q = codebook[:, idx]^T  (straight-through forward = q)
  for (int idx = tid; idx < BT * DIM_; idx += 256) {
    int i = idx >> 6, d = idx & 63;
    sA[idx] = cbT[(size_t)qIdx[i] * DIM_ + d];
  }
  __syncthreads();

  dense_bank(sA, sB, sW, wt4 + (size_t)f * 256 * 64, b4 + (size_t)f * U1_,
             64, U1_, 256, 0, false, nullptr, 0);
  __syncthreads();
  dense_bank(sB, sA, sW, wt5 + (size_t)f * 512 * 256, b5 + (size_t)f * U0_,
             256, U0_, 512, 0, false, nullptr, 0);
  __syncthreads();
  float* goutRow0 = out + ((size_t)bbase * F_ + f) * FTS_;
  dense_bank(sA, nullptr, sW, wt6 + (size_t)f * 256 * 512, b6 + (size_t)f * FTS_,
             512, FTS_, 256, 1, true, goutRow0, F_ * FTS_);
}

// ---------------------------------------------------------------------------
extern "C" void kernel_launch(void* const* d_in, const int* in_sizes, int n_in,
                              void* d_out, int out_size, void* d_ws, size_t ws_size,
                              hipStream_t stream) {
  (void)in_sizes; (void)n_in; (void)out_size; (void)ws_size;
  const float* x  = (const float*)d_in[0];
  const float* w1 = (const float*)d_in[1];
  const float* b1 = (const float*)d_in[2];
  const float* w2 = (const float*)d_in[3];
  const float* b2 = (const float*)d_in[4];
  const float* w3 = (const float*)d_in[5];
  const float* b3 = (const float*)d_in[6];
  const float* cb = (const float*)d_in[7];
  const float* w4 = (const float*)d_in[8];
  const float* b4 = (const float*)d_in[9];
  const float* w5 = (const float*)d_in[10];
  const float* b5 = (const float*)d_in[11];
  const float* w6 = (const float*)d_in[12];
  const float* b6 = (const float*)d_in[13];
  float* out = (float*)d_out;

  // Workspace layout: six bf16 [F][Opad][Kpad] weight banks + bf16 codebook^T + cnorm
  // (~285.3 MB total)
  unsigned short* ws = (unsigned short*)d_ws;
  const size_t e1 = (size_t)F_ * 512 * 256, e2 = (size_t)F_ * 256 * 512,
               e3 = (size_t)F_ * 64 * 256,  e4 = (size_t)F_ * 256 * 64,
               e5 = (size_t)F_ * 512 * 256, e6 = (size_t)F_ * 256 * 512,
               ecb = (size_t)EMB_ * DIM_;
  unsigned short* wt1 = ws;        unsigned short* wt2 = wt1 + e1;
  unsigned short* wt3 = wt2 + e2;  unsigned short* wt4 = wt3 + e3;
  unsigned short* wt5 = wt4 + e4;  unsigned short* wt6 = wt5 + e5;
  unsigned short* cbT = wt6 + e6;
  float* cnorm = (float*)(cbT + ecb);

  auto launch_w = [&](const float* w, unsigned short* dst, int I, int O,
                      int Kpad, int Opad) {
    long total  = (long)F_ * Opad * Kpad;
    int  blocks = (int)((total + 255) / 256);
    prep_w<<<blocks, 256, 0, stream>>>(w, dst, I, O, Kpad, Opad, total);
  };
  launch_w(w1, wt1, FTS_, U0_, 256, 512);
  launch_w(w2, wt2, U0_, U1_, 512, 256);
  launch_w(w3, wt3, U1_, DIM_, 256, 64);
  launch_w(w4, wt4, DIM_, U1_, 64, 256);
  launch_w(w5, wt5, U1_, U0_, 256, 512);
  launch_w(w6, wt6, U0_, FTS_, 512, 256);
  prep_cb<<<2, 256, 0, stream>>>(cb, cbT, cnorm);

  vqvae_fused<<<F_ * (B_ / BT), 256, 0, stream>>>(
      x, b1, b2, b3, b4, b5, b6,
      wt1, wt2, wt3, wt4, wt5, wt6, cbT, cnorm, out);
}
